// sparse_attn_61701500175305
// MI455X (gfx1250) — matrix operand order repaired and checked by reading, not compiled, not run on hardware
//
#include <hip/hip_runtime.h>
#include <math.h>

typedef __attribute__((ext_vector_type(16))) _Float16 v16h;
typedef __attribute__((ext_vector_type(8)))  float    v8f;

#define WMMA_F32_F16(a, b, c) \
  __builtin_amdgcn_wmma_f32_16x16x32_f16(false, (a), false, (b), (short)0, (c), false, false)

// ---------------------------------------------------------------------------
// WMMA fragment loaders (contiguous k order: k = 16*(lane>>4) + s, wave32)
// ---------------------------------------------------------------------------

// A fragment: 16x32 f16, rows m0..m0+15, cols k0..k0+31, from row-major f32.
// lane&15 -> M row; lane>=16 selects K+16; regs pack K pairs.
__device__ __forceinline__ v16h load_a_frag(const float* A, int lda, int m0, int k0, float scale) {
  const int lane = threadIdx.x & 31;
  const float* row = A + (long)(m0 + (lane & 15)) * lda + k0;
  const int koff = (lane >> 4) << 4;
  v16h a;
#pragma unroll
  for (int r = 0; r < 8; ++r) {
    int kb = 2 * r + koff;
    a[2 * r]     = (_Float16)(row[kb] * scale);
    a[2 * r + 1] = (_Float16)(row[kb + 1] * scale);
  }
  return a;
}

// B fragment: 32x16 f16 where logical B[k][n] = W[n*ldw + k]  (B = W^T, W row-major [N,K]).
// lane&15 -> N col; lane>=16 selects K+16; regs pack K pairs.
__device__ __forceinline__ v16h load_b_frag_wt(const float* W, int ldw, int k0, int n0) {
  const int lane = threadIdx.x & 31;
  const float* row = W + (long)(n0 + (lane & 15)) * ldw + k0 + ((lane >> 4) << 4);
  v16h b;
#pragma unroll
  for (int r = 0; r < 8; ++r) {
    b[2 * r]     = (_Float16)row[2 * r];
    b[2 * r + 1] = (_Float16)row[2 * r + 1];
  }
  return b;
}

// B fragment from row-major B[k][n] (ldb = row stride).
__device__ __forceinline__ v16h load_b_frag_rm(const float* B, int ldb, int k0, int n0) {
  const int lane = threadIdx.x & 31;
  const float* p = B + (long)(k0 + ((lane >> 4) << 4)) * ldb + n0 + (lane & 15);
  v16h b;
#pragma unroll
  for (int r = 0; r < 8; ++r) {
    b[2 * r]     = (_Float16)p[(2 * r) * (long)ldb];
    b[2 * r + 1] = (_Float16)p[(2 * r + 1) * (long)ldb];
  }
  return b;
}

// ---------------------------------------------------------------------------
// Generic GEMM: C[M,N] = act(A[M,K] @ W[N,K]^T + bias), 1 wave -> 32x32 tile
// ---------------------------------------------------------------------------
__global__ void gemm_bias_act(const float* __restrict__ A, const float* __restrict__ W,
                              const float* __restrict__ bias, float* __restrict__ C,
                              int M, int N, int K, int relu) {
  const int ntiles = N >> 5;
  const int tm = blockIdx.x / ntiles, tn = blockIdx.x % ntiles;
  const int m0 = tm << 5, n0 = tn << 5;
  const int lane = threadIdx.x & 31;
  v8f acc[2][2] = {};
  for (int k0 = 0; k0 < K; k0 += 32) {
    v16h a0 = load_a_frag(A, K, m0, k0, 1.0f);
    v16h a1 = load_a_frag(A, K, m0 + 16, k0, 1.0f);
    v16h b0 = load_b_frag_wt(W, K, k0, n0);
    v16h b1 = load_b_frag_wt(W, K, k0, n0 + 16);
    acc[0][0] = WMMA_F32_F16(a0, b0, acc[0][0]);
    acc[0][1] = WMMA_F32_F16(a0, b1, acc[0][1]);
    acc[1][0] = WMMA_F32_F16(a1, b0, acc[1][0]);
    acc[1][1] = WMMA_F32_F16(a1, b1, acc[1][1]);
  }
  const int mo = (lane >> 4) << 3;
#pragma unroll
  for (int i = 0; i < 2; ++i)
#pragma unroll
    for (int j = 0; j < 2; ++j) {
      int n = n0 + (j << 4) + (lane & 15);
      float bv = bias[n];
#pragma unroll
      for (int r = 0; r < 8; ++r) {
        float v = acc[i][j][r] + bv;
        if (relu) v = fmaxf(v, 0.0f);
        C[(long)(m0 + (i << 4) + mo + r) * N + n] = v;
      }
    }
}

// ---------------------------------------------------------------------------
// Fused flash attention: 1 wave handles 16 queries of one head.
// qkv: [L, 768] f32 (q|k|v each 256 wide, head dim 32).  u: [L,L] f32.
// mask(q,k) = (u>=0.1) + (pos_q==pos_k) + (frame_q==frame_k), added pre-softmax.
// ---------------------------------------------------------------------------
__global__ void flash_attn(const float* __restrict__ qkv, const float* __restrict__ u,
                           float* __restrict__ O, int L, int S) {
  __shared__ float stile[16 * 32];
  __shared__ float alphaS[16];
  __shared__ float lS[16];
  const int lane = threadIdx.x & 31;
  const int head = blockIdx.y;
  const int q0 = blockIdx.x << 4;
  const int mo = (lane >> 4) << 3;
  const int nl = lane & 15;
  const float scale = 0.17677669529663687f;  // 1/sqrt(32)

  v16h qfrag = load_a_frag(qkv + head * 32, 768, q0, 0, scale);
  v8f acc0 = {}, acc1 = {};
  float mrun = -1e30f, lrun = 0.0f;  // owned by lanes 0..15 (row = lane)

  for (int kc = 0; kc < L; kc += 32) {
    // S tile = Q @ K^T : K^T[d][key] = qkv[key*768 + 256 + head*32 + d]
    v16h bk0 = load_b_frag_wt(qkv + 256 + head * 32, 768, 0, kc);
    v16h bk1 = load_b_frag_wt(qkv + 256 + head * 32, 768, 0, kc + 16);
    v8f s0 = {}, s1 = {};
    s0 = WMMA_F32_F16(qfrag, bk0, s0);
    s1 = WMMA_F32_F16(qfrag, bk1, s1);

    // add mask, dump D-layout tile to LDS
#pragma unroll
    for (int r = 0; r < 8; ++r) {
      int qq = q0 + mo + r;
      int fq = qq / S, pq = qq % S;
      int k0g = kc + nl, k1g = kc + 16 + nl;
      float m0v = (u[(long)qq * L + k0g] >= 0.1f ? 1.f : 0.f) +
                  (pq == (k0g % S) ? 1.f : 0.f) + (fq == (k0g / S) ? 1.f : 0.f);
      float m1v = (u[(long)qq * L + k1g] >= 0.1f ? 1.f : 0.f) +
                  (pq == (k1g % S) ? 1.f : 0.f) + (fq == (k1g / S) ? 1.f : 0.f);
      stile[(mo + r) * 32 + nl]      = s0[r] + m0v;
      stile[(mo + r) * 32 + 16 + nl] = s1[r] + m1v;
    }
    __syncthreads();

    // online softmax row update (lane i owns row i)
    if (lane < 16) {
      float* row = stile + lane * 32;
      float cmax = row[0];
#pragma unroll
      for (int t = 1; t < 32; ++t) cmax = fmaxf(cmax, row[t]);
      float newm = fmaxf(mrun, cmax);
      float alpha = __expf(mrun - newm);
      float ssum = 0.f;
#pragma unroll
      for (int t = 0; t < 32; ++t) {
        float p = __expf(row[t] - newm);
        row[t] = p;
        ssum += p;
      }
      lrun = lrun * alpha + ssum;
      mrun = newm;
      alphaS[lane] = alpha;
    }
    __syncthreads();

    // rescale accumulators, then O += P @ V
#pragma unroll
    for (int r = 0; r < 8; ++r) {
      float al = alphaS[mo + r];
      acc0[r] *= al;
      acc1[r] *= al;
    }
    v16h pfrag = load_a_frag(stile, 32, 0, 0, 1.0f);
    v16h bv0 = load_b_frag_rm(qkv + 512 + head * 32, 768, kc, 0);
    v16h bv1 = load_b_frag_rm(qkv + 512 + head * 32, 768, kc, 16);
    acc0 = WMMA_F32_F16(pfrag, bv0, acc0);
    acc1 = WMMA_F32_F16(pfrag, bv1, acc1);
    __syncthreads();
  }

  if (lane < 16) lS[lane] = lrun;
  __syncthreads();
#pragma unroll
  for (int r = 0; r < 8; ++r) {
    float inv = 1.0f / lS[mo + r];
    int qq = q0 + mo + r;
    O[(long)qq * 256 + head * 32 + nl]      = acc0[r] * inv;
    O[(long)qq * 256 + head * 32 + 16 + nl] = acc1[r] * inv;
  }
}

// ---------------------------------------------------------------------------
// out = LayerNorm(A + B) * g + beta   (one 256-thread block per row, C == 256)
// ---------------------------------------------------------------------------
__global__ void add_ln(const float* __restrict__ A, const float* __restrict__ B,
                       const float* __restrict__ g, const float* __restrict__ be,
                       float* __restrict__ out, int C) {
  __shared__ float red[256];
  const int row = blockIdx.x, c = threadIdx.x;
  float v = A[(long)row * C + c] + B[(long)row * C + c];
  red[c] = v;
  __syncthreads();
  for (int s = 128; s > 0; s >>= 1) {
    if (c < s) red[c] += red[c + s];
    __syncthreads();
  }
  float mean = red[0] / C;
  __syncthreads();
  float d = v - mean;
  red[c] = d * d;
  __syncthreads();
  for (int s = 128; s > 0; s >>= 1) {
    if (c < s) red[c] += red[c + s];
    __syncthreads();
  }
  float var = red[0] / C;
  out[(long)row * C + c] = d * rsqrtf(var + 1e-5f) * g[c] + be[c];
}

// NCHW -> [L, C] token rows (l = (n*H + h)*W + w)
__global__ void to_rows(const float* __restrict__ f, float* __restrict__ src,
                        int HW, int C, int total) {
  int i = blockIdx.x * blockDim.x + threadIdx.x;
  if (i >= total) return;
  int c = i % C, l = i / C;
  int n = l / HW, hw = l % HW;
  src[i] = f[((long)n * C + c) * HW + hw];
}

// [L, C] token rows -> NCHW
__global__ void from_rows(const float* __restrict__ x, float* __restrict__ out,
                          int HW, int C, int total) {
  int i = blockIdx.x * blockDim.x + threadIdx.x;
  if (i >= total) return;
  int n = i / (C * HW);
  int rem = i % (C * HW);
  int c = rem / HW, hw = rem % HW;
  out[i] = x[(long)(n * HW + hw) * C + c];
}

// ---------------------------------------------------------------------------
extern "C" void kernel_launch(void* const* d_in, const int* in_sizes, int n_in,
                              void* d_out, int out_size, void* d_ws, size_t ws_size,
                              hipStream_t stream) {
  float* out = (float*)d_out;

  // levels 0 and 1 pass through untouched
  hipMemcpyAsync(out, d_in[0], (size_t)in_sizes[0] * sizeof(float),
                 hipMemcpyDeviceToDevice, stream);
  hipMemcpyAsync(out + in_sizes[0], d_in[1], (size_t)in_sizes[1] * sizeof(float),
                 hipMemcpyDeviceToDevice, stream);

  const float* Wqkv = (const float*)d_in[6];
  const float* bqkv = (const float*)d_in[7];
  const float* Wo   = (const float*)d_in[8];
  const float* bo   = (const float*)d_in[9];
  const float* g1   = (const float*)d_in[10];
  const float* b1   = (const float*)d_in[11];
  const float* Wfc1 = (const float*)d_in[12];
  const float* bfc1 = (const float*)d_in[13];
  const float* Wfc2 = (const float*)d_in[14];
  const float* bfc2 = (const float*)d_in[15];
  const float* g2   = (const float*)d_in[16];
  const float* b2   = (const float*)d_in[17];

  // workspace layout (sized for L = 2048; ~27 MB total)
  float* src  = (float*)d_ws;            // [L,256]
  float* qkv  = src  + 2048 * 256;       // [L,768]
  float* attn = qkv  + 2048 * 768;       // [L,256]
  float* src2 = attn + 2048 * 256;       // [L,256]
  float* x1   = src2 + 2048 * 256;       // [L,256]
  float* hbuf = x1   + 2048 * 256;       // [L,1024]
  float* x2   = hbuf + 2048 * 1024;      // [L,256]
  float* xf   = x2   + 2048 * 256;       // [L,256]

  size_t outoff = (size_t)in_sizes[0] + in_sizes[1];
  for (int lvl = 0; lvl < 2; ++lvl) {
    const float* f = (const float*)d_in[2 + lvl];
    const float* u = (const float*)d_in[4 + lvl];
    const int HW = (lvl == 0) ? 1024 : 256;  // 32x32, 16x16
    const int L = 2 * HW;
    const int total = L * 256;

    to_rows<<<(total + 255) / 256, 256, 0, stream>>>(f, src, HW, 256, total);
    gemm_bias_act<<<(L / 32) * (768 / 32), 32, 0, stream>>>(src, Wqkv, bqkv, qkv,
                                                            L, 768, 256, 0);
    flash_attn<<<dim3(L / 16, 8), 32, 0, stream>>>(qkv, u, attn, L, HW);
    gemm_bias_act<<<(L / 32) * (256 / 32), 32, 0, stream>>>(attn, Wo, bo, src2,
                                                            L, 256, 256, 0);
    add_ln<<<L, 256, 0, stream>>>(src, src2, g1, b1, x1, 256);
    gemm_bias_act<<<(L / 32) * (1024 / 32), 32, 0, stream>>>(x1, Wfc1, bfc1, hbuf,
                                                             L, 1024, 256, 1);
    gemm_bias_act<<<(L / 32) * (256 / 32), 32, 0, stream>>>(hbuf, Wfc2, bfc2, x2,
                                                            L, 256, 1024, 0);
    add_ln<<<L, 256, 0, stream>>>(x1, x2, g2, b2, xf, 256);
    from_rows<<<(total + 255) / 256, 256, 0, stream>>>(xf, out + outoff, HW, 256, total);
    outoff += total;
  }
}
